// WindowAttention_34634616275339
// MI455X (gfx1250) — compile-verified, matrix operand order checked by reading, not run on hardware
//
#include <hip/hip_runtime.h>
#include <hip/hip_bf16.h>
#include <stdint.h>

#define DIM   128
#define NHEAD 4
#define DH    32
#define NSEQ  256
#define BATCH 512
#define NWIN  64

typedef __attribute__((ext_vector_type(16))) __bf16 bf16x16;
typedef __attribute__((ext_vector_type(8)))  float  f32x8;

union Frag16 { uint4 q[2]; bf16x16 v; };

// Pack two f32 to bf16x2 (truncation) in a single v_perm_b32:
// result bytes = { b[3], b[2], a[3], a[2] }  ->  {bf16(b), bf16(a)}
__device__ __forceinline__ uint32_t pack2_bf16(float a, float b) {
  return __builtin_amdgcn_perm(__float_as_uint(b), __float_as_uint(a), 0x07060302u);
}

// Store bf16 (truncation); the >>16 folds into *_store_d16_hi_b16.
__device__ __forceinline__ void store_bf16(uint16_t* p, float x) {
  *p = (uint16_t)(__float_as_uint(x) >> 16);
}

// Load a 16x32 bf16 WMMA fragment (A layout; B operands use the same pattern
// since all our B matrices are stored N-major with K contiguous).
// lane<16:  row=lane,    K = {0..7, 16..23}
// lane>=16: row=lane-16, K = {8..15, 24..31}
__device__ __forceinline__ bf16x16 load_frag16(const uint16_t* base, int row0, int stride) {
  const int lane = threadIdx.x & 31;
  const int half = lane >> 4, r = lane & 15;
  const uint16_t* p = base + (size_t)(row0 + r) * stride + half * 8;
  Frag16 f;
  f.q[0] = *(const uint4*)(p);
  f.q[1] = *(const uint4*)(p + 16);
  return f.v;
}

// Same fragment but sourced from fp32 memory with on-the-fly bf16 conversion.
__device__ __forceinline__ bf16x16 load_frag16_f32(const float* base, int row0, int stride) {
  const int lane = threadIdx.x & 31;
  const int half = lane >> 4, r = lane & 15;
  const float* p = base + (size_t)(row0 + r) * stride + half * 8;
  float4 a0 = *(const float4*)(p);
  float4 a1 = *(const float4*)(p + 4);
  float4 b0 = *(const float4*)(p + 16);
  float4 b1 = *(const float4*)(p + 20);
  Frag16 f;
  f.q[0] = make_uint4(pack2_bf16(a0.x, a0.y), pack2_bf16(a0.z, a0.w),
                      pack2_bf16(a1.x, a1.y), pack2_bf16(a1.z, a1.w));
  f.q[1] = make_uint4(pack2_bf16(b0.x, b0.y), pack2_bf16(b0.z, b0.w),
                      pack2_bf16(b1.x, b1.y), pack2_bf16(b1.z, b1.w));
  return f.v;
}

__device__ __forceinline__ f32x8 wmma_bf16(bf16x16 a, bf16x16 b, f32x8 c) {
  return __builtin_amdgcn_wmma_f32_16x16x32_bf16(false, a, false, b, (short)0, c,
                                                 false, false);
}

// ---------------------------------------------------------------------------
// Kernel 0: rpb[h][n][m] = bias_table[rel_idx[n][m]][h]
// ---------------------------------------------------------------------------
__global__ void rpb_kernel(const float* __restrict__ bias_table,
                           const int* __restrict__ rel_idx,
                           float* __restrict__ rpb) {
  const int n = blockIdx.x, m = threadIdx.x;
  const int idx = rel_idx[n * NSEQ + m];
  #pragma unroll
  for (int h = 0; h < NHEAD; ++h)
    rpb[((size_t)h * NSEQ + n) * NSEQ + m] = bias_table[idx * NHEAD + h];
}

// ---------------------------------------------------------------------------
// Kernel 1: QKV = X @ Wqkv^T + b  -> bf16 Q(scaled)/K/V as [b][h][n][dh]
// block = 256 thr (8 waves), 128 rows/block, full 384 cols, K=128.
// nt loop fully unrolled: which/h are compile-time constants per tile.
// ---------------------------------------------------------------------------
__global__ void __launch_bounds__(256) qkv_kernel(
    const float* __restrict__ X, const float* __restrict__ Wqkv,
    const float* __restrict__ bqkv,
    uint16_t* __restrict__ Q, uint16_t* __restrict__ Kt,
    uint16_t* __restrict__ V) {
  __shared__ uint16_t Wl[384 * DIM];  // bf16 [384][128], 96 KB
  {
    const float2* src = (const float2*)Wqkv;
    uint32_t* dst = (uint32_t*)Wl;
    for (int i = threadIdx.x; i < 384 * DIM / 2; i += 256) {
      float2 f = src[i];
      dst[i] = pack2_bf16(f.x, f.y);
    }
  }
  __syncthreads();

  const int wid  = threadIdx.x >> 5;
  const int lane = threadIdx.x & 31;
  const int half = lane >> 4;
  const int m0   = blockIdx.x * 128 + wid * 16;
  const int b_   = m0 >> 8;                 // window index (tile never crosses)
  const int n0   = (m0 & 255) + 8 * half;   // first row handled by this half

  bf16x16 afrag[4];
  #pragma unroll
  for (int kc = 0; kc < 4; ++kc)
    afrag[kc] = load_frag16_f32(X + kc * 32, m0, DIM);

  #pragma unroll
  for (int nt = 0; nt < 24; ++nt) {
    f32x8 acc = {};
    #pragma unroll
    for (int kc = 0; kc < 4; ++kc) {
      bf16x16 bfr = load_frag16(Wl + kc * 32, nt * 16, DIM);
      acc = wmma_bf16(afrag[kc], bfr, acc);
    }
    const int c     = nt * 16 + (lane & 15);
    const int which = nt >> 3;             // compile-time: 0=q, 1=k, 2=v
    const int h     = (nt & 7) >> 1;       // compile-time head
    const int dh    = c & 31;
    const float bias = bqkv[c];
    const float mul  = (which == 0) ? 0.17677669529663687f : 1.0f;
    uint16_t* dstp = (which == 0) ? Q : ((which == 1) ? Kt : V);
    uint16_t* p = dstp + ((size_t)(b_ * NHEAD + h) * NSEQ + n0) * DH + dh;
    const float* af = (const float*)&acc;
    #pragma unroll
    for (int r = 0; r < 8; ++r)
      store_bf16(p + r * DH, (af[r] + bias) * mul);
  }
}

// ---------------------------------------------------------------------------
// Kernel 2: attention per (b,h). block = 256 thr (8 waves), 2 q-tiles/wave.
// S (16x256 f32) kept in WMMA accumulators; softmax in-register + shfl;
// P staged bf16 in LDS and re-read in A layout for P@V.
// ---------------------------------------------------------------------------
__global__ void __launch_bounds__(256) attn_kernel(
    const uint16_t* __restrict__ Q, const uint16_t* __restrict__ K,
    const uint16_t* __restrict__ V, const float* __restrict__ rpb,
    const float* __restrict__ mask, uint16_t* __restrict__ O) {
  __shared__ uint16_t Kl[NSEQ * DH];       // [n][dh]     16 KB
  __shared__ uint16_t Vtl[DH * NSEQ];      // [dh][n]     16 KB
  __shared__ uint16_t Pl[8 * 16 * NSEQ];   // per-wave P  64 KB

  const int bh = blockIdx.x;
  const int b = bh >> 2, h = bh & 3;
  const int w = b & (NWIN - 1);
  const size_t base = (size_t)bh * NSEQ * DH;

  {  // K -> LDS (linear)
    const uint4* src = (const uint4*)(K + base);
    uint4* dst = (uint4*)Kl;
    for (int i = threadIdx.x; i < NSEQ * DH / 8; i += 256) dst[i] = src[i];
  }
  {  // V -> LDS transposed: Vtl[dh][n]
    const int n = threadIdx.x;
    const uint32_t* vp = (const uint32_t*)(V + base + (size_t)n * DH);
    #pragma unroll
    for (int j = 0; j < DH / 2; ++j) {
      const uint32_t u = vp[j];
      Vtl[(2 * j) * NSEQ + n]     = (uint16_t)u;
      Vtl[(2 * j + 1) * NSEQ + n] = (uint16_t)(u >> 16);
    }
  }
  __syncthreads();

  const int wid  = threadIdx.x >> 5;
  const int lane = threadIdx.x & 31;
  const int half = lane >> 4;
  const int mcol = lane & 15;
  uint16_t* Pw = Pl + wid * 16 * NSEQ;

  for (int qt = 0; qt < 2; ++qt) {
    const int q0 = (wid * 2 + qt) * 16;
    const bf16x16 qa = load_frag16(Q + base, q0, DH);

    // S = q @ k^T : 16 WMMAs (K=Dh=32 -> single k-chunk each)
    f32x8 sacc[16];
    #pragma unroll
    for (int t = 0; t < 16; ++t) {
      bf16x16 kb = load_frag16(Kl, t * 16, DH);
      f32x8 z = {};
      sacc[t] = wmma_bf16(qa, kb, z);
    }
    float* sf = (float*)sacc;

    // + relative-position bias + window mask
    #pragma unroll
    for (int t = 0; t < 16; ++t) {
      const int m = t * 16 + mcol;
      #pragma unroll
      for (int r = 0; r < 8; ++r) {
        const int n = q0 + r + 8 * half;
        sf[t * 8 + r] += rpb[((size_t)h * NSEQ + n) * NSEQ + m] +
                         mask[((size_t)w * NSEQ + n) * NSEQ + m];
      }
    }

    // softmax: row max / exp / row sum (rows live in half-wave lane groups)
    float rmax[8], rsum[8];
    #pragma unroll
    for (int r = 0; r < 8; ++r) {
      float mx = sf[r];
      #pragma unroll
      for (int t = 1; t < 16; ++t) mx = fmaxf(mx, sf[t * 8 + r]);
      #pragma unroll
      for (int s = 1; s < 16; s <<= 1) mx = fmaxf(mx, __shfl_xor(mx, s, 32));
      rmax[r] = mx;
    }
    #pragma unroll
    for (int r = 0; r < 8; ++r) {
      float sum = 0.f;
      #pragma unroll
      for (int t = 0; t < 16; ++t) {
        const float p = __expf(sf[t * 8 + r] - rmax[r]);
        sf[t * 8 + r] = p;
        sum += p;
      }
      #pragma unroll
      for (int s = 1; s < 16; s <<= 1) sum += __shfl_xor(sum, s, 32);
      rsum[r] = sum;
    }

    // P -> LDS as bf16 row-major [16][256] (C-layout -> A-layout transpose)
    #pragma unroll
    for (int t = 0; t < 16; ++t) {
      const int m = t * 16 + mcol;
      #pragma unroll
      for (int r = 0; r < 8; ++r)
        store_bf16(&Pw[(r + 8 * half) * NSEQ + m], sf[t * 8 + r]);
    }

    // O = P @ V : 8 k-chunks x 2 n-tiles = 16 WMMAs
    f32x8 o0 = {}, o1 = {};
    #pragma unroll
    for (int kc = 0; kc < 8; ++kc) {
      bf16x16 pa = load_frag16(Pw + kc * 32, 0, NSEQ);
      bf16x16 v0 = load_frag16(Vtl + kc * 32, 0, NSEQ);
      bf16x16 v1 = load_frag16(Vtl + kc * 32, 16, NSEQ);
      o0 = wmma_bf16(pa, v0, o0);
      o1 = wmma_bf16(pa, v1, o1);
    }
    const float* of0 = (const float*)&o0;
    const float* of1 = (const float*)&o1;
    #pragma unroll
    for (int r = 0; r < 8; ++r) {
      const int n = q0 + r + 8 * half;
      const float inv = 1.0f / rsum[r];
      const size_t dst = ((size_t)b * NSEQ + n) * DIM + h * DH;
      store_bf16((uint16_t*)O + dst + mcol,      of0[r] * inv);
      store_bf16((uint16_t*)O + dst + 16 + mcol, of1[r] * inv);
    }
  }
}

// ---------------------------------------------------------------------------
// Kernel 3: Y = O @ proj_w^T + proj_b  (fp32 out)
// ---------------------------------------------------------------------------
__global__ void __launch_bounds__(256) proj_kernel(
    const uint16_t* __restrict__ O, const float* __restrict__ Wp,
    const float* __restrict__ bp, float* __restrict__ Y) {
  __shared__ uint16_t Wl[DIM * DIM];  // bf16 [128][128], 32 KB
  {
    const float2* src = (const float2*)Wp;
    uint32_t* dst = (uint32_t*)Wl;
    for (int i = threadIdx.x; i < DIM * DIM / 2; i += 256) {
      float2 f = src[i];
      dst[i] = pack2_bf16(f.x, f.y);
    }
  }
  __syncthreads();

  const int wid  = threadIdx.x >> 5;
  const int lane = threadIdx.x & 31;
  const int half = lane >> 4;
  const int m0   = blockIdx.x * 128 + wid * 16;

  bf16x16 afrag[4];
  #pragma unroll
  for (int kc = 0; kc < 4; ++kc)
    afrag[kc] = load_frag16(O + kc * 32, m0, DIM);

  #pragma unroll
  for (int nt = 0; nt < 8; ++nt) {
    f32x8 acc = {};
    #pragma unroll
    for (int kc = 0; kc < 4; ++kc) {
      bf16x16 bfr = load_frag16(Wl + kc * 32, nt * 16, DIM);
      acc = wmma_bf16(afrag[kc], bfr, acc);
    }
    const int c = nt * 16 + (lane & 15);
    const float bias = bp[c];
    const float* af = (const float*)&acc;
    #pragma unroll
    for (int r = 0; r < 8; ++r) {
      const int M = m0 + r + 8 * half;
      Y[(size_t)M * DIM + c] = af[r] + bias;
    }
  }
}

// ---------------------------------------------------------------------------
extern "C" void kernel_launch(void* const* d_in, const int* in_sizes, int n_in,
                              void* d_out, int out_size, void* d_ws, size_t ws_size,
                              hipStream_t stream) {
  const float* X          = (const float*)d_in[0];
  const float* mask       = (const float*)d_in[1];
  const float* qkv_w      = (const float*)d_in[2];
  const float* qkv_b      = (const float*)d_in[3];
  const float* proj_w     = (const float*)d_in[4];
  const float* proj_b     = (const float*)d_in[5];
  const float* bias_table = (const float*)d_in[6];
  const int*   rel_idx    = (const int*)d_in[7];

  char* ws = (char*)d_ws;
  const size_t qkv_elems = (size_t)BATCH * NHEAD * NSEQ * DH;  // 16.78M
  float*    rpb = (float*)ws;                                  // 1 MB
  uint16_t* Q   = (uint16_t*)(ws + (1u << 20));
  uint16_t* K   = Q + qkv_elems;
  uint16_t* V   = K + qkv_elems;
  uint16_t* O   = V + qkv_elems;                               // [b][n][128] bf16

  hipLaunchKernelGGL(rpb_kernel, dim3(NSEQ), dim3(NSEQ), 0, stream,
                     bias_table, rel_idx, rpb);
  hipLaunchKernelGGL(qkv_kernel, dim3(BATCH * NSEQ / 128), dim3(256), 0, stream,
                     X, qkv_w, qkv_b, Q, K, V);
  hipLaunchKernelGGL(attn_kernel, dim3(BATCH * NHEAD), dim3(256), 0, stream,
                     Q, K, V, rpb, mask, O);
  hipLaunchKernelGGL(proj_kernel, dim3(BATCH * NSEQ / 128), dim3(256), 0, stream,
                     O, proj_w, proj_b, (float*)d_out);
}
